// RNN_86208583565808
// MI455X (gfx1250) — matrix operand order repaired and checked by reading, not compiled, not run on hardware
//
#include <hip/hip_runtime.h>
#include <hip/hip_bf16.h>
#include <stdint.h>

#define T_STEPS 1000
#define BATCH   128
#define IDIM    36
#define HDIM    512
#define ALPHA_F 0.2f
#define SIGMA_F 0.15811388300841897f   // sqrt(2/0.2)*0.05

typedef __attribute__((ext_vector_type(16))) __bf16 v16bf;
typedef __attribute__((ext_vector_type(8)))  __bf16 v8bf;
typedef __attribute__((ext_vector_type(8)))  float  v8f;

// LDS map (dynamic): [0, 16KB) = bf16 A matrix (16 x 512)
//                    [16KB, 272KB) = weight fragments, 1KB each:
//                    frag id = ((wave*8 + kstep)*2 + tile), lane's 32B at lane*32
#define A_BYTES    (16 * HDIM * 2)          // 16384
#define W_BYTES    (16 * 8 * 2 * 1024)      // 262144
#define LDS_BYTES  (A_BYTES + W_BYTES)      // 278528

__device__ __forceinline__ __bf16 f2bf(float x) {
    union { float f; unsigned u; } v; v.f = x;
    unsigned r = (v.u + 0x7FFFu + ((v.u >> 16) & 1u)) >> 16;
    union { unsigned short s; __bf16 b; } o; o.s = (unsigned short)r;
    return o.b;
}

// ---------------------------------------------------------------------------
// Kernel 0: weight_hh (f32, [k][n]) -> bf16 transposed [n][k] in workspace.
// ---------------------------------------------------------------------------
__global__ void wconv_kernel(const float* __restrict__ whh, __bf16* __restrict__ wT) {
    int n = blockIdx.x;
    int k = threadIdx.x;
    wT[(size_t)n * HDIM + k] = f2bf(whh[(size_t)k * HDIM + n]);
}

// ---------------------------------------------------------------------------
// Kernel 1: ext[t,b,h] = inputs @ weight_ih + bias + SIGMA*noise, written to
// d_out[(t+1)*B*H + b*H + h]; scan overwrites in place. Bandwidth bound.
// ---------------------------------------------------------------------------
__global__ __launch_bounds__(HDIM) void ext_kernel(const float* __restrict__ inputs,
                                                   const float* __restrict__ w_ih,
                                                   const float* __restrict__ bias,
                                                   const float* __restrict__ noise,
                                                   float* __restrict__ out) {
    const int tb = blockIdx.x;         // t*B + b
    const int h  = threadIdx.x;        // 0..511
    __shared__ float xin[IDIM];
    if (h < IDIM) xin[h] = inputs[(size_t)tb * IDIM + h];
    __syncthreads();
    float acc = bias[h] + SIGMA_F * noise[(size_t)tb * HDIM + h];
#pragma unroll
    for (int i = 0; i < IDIM; ++i)
        acc = fmaf(xin[i], w_ih[(size_t)i * HDIM + h], acc);
    out[(size_t)(tb + BATCH) * HDIM + h] = acc;    // (t+1)*B*H + b*H == (tb+B)*H
}

// ---------------------------------------------------------------------------
// Kernel 2: persistent scan. grid = 8 batch groups x 512 threads (16 waves).
// Wave w owns N columns [32w, 32w+32). Weights fully resident:
//   k in [0,256)   : LDS   (async-copied once, ds_load_b128 per step)
//   k in [256,512) : VGPRs (128 regs/lane, loaded once)
// ---------------------------------------------------------------------------
__global__ __launch_bounds__(512, 1) void scan_kernel(const float* __restrict__ init,
                                                      const __bf16* __restrict__ wT,
                                                      float* __restrict__ out) {
    extern __shared__ char lds_buf[];
    __bf16* sA = (__bf16*)lds_buf;

    const int lane = threadIdx.x & 31;
    const int wv   = threadIdx.x >> 5;   // 0..15
    const int hi   = lane >> 4;          // 0/1
    const int lo   = lane & 15;
    const int n0   = wv * 32;
    const int Mrow = lo;

    const size_t slice = (size_t)BATCH * HDIM;            // 65536
    const size_t base  = (size_t)blockIdx.x * 16 * HDIM;  // batch-group offset

    // ---- async-copy low-K weight fragments into this wave's LDS region ----
#pragma unroll
    for (int ks = 0; ks < 8; ++ks) {
#pragma unroll
        for (int t2 = 0; t2 < 2; ++t2) {
            const int frag = (wv * 8 + ks) * 2 + t2;
            const unsigned lds0 = (unsigned)(A_BYTES + frag * 1024 + lane * 32);
            const __bf16* gp = &wT[(size_t)(n0 + t2 * 16 + lo) * HDIM + ks * 32 + hi * 16];
            const uint64_t ga = (uint64_t)(uintptr_t)gp;
            // per ISA 08_async_tensor: INST_OFFSET applies to both LDS and global
            asm volatile("global_load_async_to_lds_b128 %0, %1, off"
                         :: "v"(lds0), "v"(ga) : "memory");
            asm volatile("global_load_async_to_lds_b128 %0, %1, off offset:16"
                         :: "v"(lds0), "v"(ga) : "memory");
        }
    }

    // ---- high-K weight fragments pinned in registers (128 VGPRs/lane) ----
    v16bf breg[8][2];
#pragma unroll
    for (int ks = 0; ks < 8; ++ks)
#pragma unroll
        for (int t2 = 0; t2 < 2; ++t2)
            breg[ks][t2] = *(const v16bf*)
                (&wT[(size_t)(n0 + t2 * 16 + lo) * HDIM + 256 + ks * 32 + hi * 16]);

    // ---- initial state into fragment-layout registers, emit out[0] ----
    float st[2][8];
#pragma unroll
    for (int t2 = 0; t2 < 2; ++t2) {
        const int n = n0 + t2 * 16 + lo;
#pragma unroll
        for (int j = 0; j < 8; ++j) {
            const int M = j + 8 * hi;
            const float v = init[base + (size_t)M * HDIM + n];
            st[t2][j] = v;
            out[base + (size_t)M * HDIM + n] = v;
        }
    }

    asm volatile("s_wait_asynccnt 0x0" ::: "memory");
    __syncthreads();

    for (int t = 0; t < T_STEPS; ++t) {
        // relu(state) -> bf16 A matrix in LDS
#pragma unroll
        for (int t2 = 0; t2 < 2; ++t2) {
#pragma unroll
            for (int j = 0; j < 8; ++j) {
                const int M = j + 8 * hi;
                const int n = n0 + t2 * 16 + lo;
                float v = st[t2][j];
                v = v > 0.0f ? v : 0.0f;
                sA[M * HDIM + n] = f2bf(v);
            }
        }
        __syncthreads();

        float* outT = out + (size_t)(t + 1) * slice;
        __builtin_prefetch(&outT[base + (size_t)(8 * hi) * HDIM + n0 + lo], 0, 3);
        __builtin_prefetch(&outT[base + (size_t)(8 * hi) * HDIM + n0 + 16 + lo], 0, 3);

        v8f acc[2] = {v8f{}, v8f{}};
#pragma unroll
        for (int ks = 0; ks < 16; ++ks) {
            const int k0 = ks * 32;
            // A fragment from LDS: lanes 0-15 K=[0..15]; 16-31 K=[16..31]
            const v8bf alo = *(const v8bf*)(&sA[Mrow * HDIM + k0 + hi * 16]);
            const v8bf ahi2 = *(const v8bf*)(&sA[Mrow * HDIM + k0 + hi * 16 + 8]);
            const v16bf a = __builtin_shufflevector(alo, ahi2,
                0, 1, 2, 3, 4, 5, 6, 7, 8, 9, 10, 11, 12, 13, 14, 15);

            v16bf b0, b1;
            if (ks < 8) {   // LDS-resident weights
                const int f0 = (wv * 8 + ks) * 2;
                b0 = *(const v16bf*)(lds_buf + A_BYTES + (f0    ) * 1024 + lane * 32);
                b1 = *(const v16bf*)(lds_buf + A_BYTES + (f0 + 1) * 1024 + lane * 32);
            } else {        // register-resident weights
                b0 = breg[ks - 8][0];
                b1 = breg[ks - 8][1];
            }
            acc[0] = __builtin_amdgcn_wmma_f32_16x16x32_bf16(
                false, a, false, b0, (short)0, acc[0], false, false);
            acc[1] = __builtin_amdgcn_wmma_f32_16x16x32_bf16(
                false, a, false, b1, (short)0, acc[1], false, false);
        }
        __syncthreads();   // all waves done reading sA before next rewrite

        // leaky update: read ext from out[t+1], overwrite with new state
#pragma unroll
        for (int t2 = 0; t2 < 2; ++t2) {
#pragma unroll
            for (int j = 0; j < 8; ++j) {
                const int M = j + 8 * hi;
                const int n = n0 + t2 * 16 + lo;
                const size_t idx = base + (size_t)M * HDIM + n;
                const float ext  = outT[idx];
                const float snew = acc[t2][j] + ext;
                const float s    = (1.0f - ALPHA_F) * st[t2][j] + ALPHA_F * snew;
                st[t2][j] = s;
                outT[idx] = s;
            }
        }
    }
}

// ---------------------------------------------------------------------------
extern "C" void kernel_launch(void* const* d_in, const int* in_sizes, int n_in,
                              void* d_out, int out_size, void* d_ws, size_t ws_size,
                              hipStream_t stream) {
    const float* inputs = (const float*)d_in[0];   // [T,B,I]
    const float* init   = (const float*)d_in[1];   // [B,H]
    const float* w_ih   = (const float*)d_in[2];   // [I,H]
    const float* w_hh   = (const float*)d_in[3];   // [H,H]
    const float* bias   = (const float*)d_in[4];   // [1,H]
    const float* noise  = (const float*)d_in[5];   // [T,B,H]
    float* out = (float*)d_out;                    // [T+1,B,H]
    __bf16* wT = (__bf16*)d_ws;                    // 512 KB bf16 weight^T

    wconv_kernel<<<HDIM, HDIM, 0, stream>>>(w_hh, wT);
    ext_kernel<<<T_STEPS * BATCH, HDIM, 0, stream>>>(inputs, w_ih, bias, noise, out);
    scan_kernel<<<BATCH / 16, 512, LDS_BYTES, stream>>>(init, wT, out);
}
